// Masked_Attention_44873818308857
// MI455X (gfx1250) — matrix operand order repaired and checked by reading, not compiled, not run on hardware
//
#include <hip/hip_runtime.h>
#include <hip/hip_bf16.h>
#include <math.h>

// ---- problem constants (reference: B=2,S=2048,E=1024,H=16,D=64) ----
#define BB 2
#define SS 2048
#define EE 1024
#define HH 16
#define DD 64
#define MM (BB * SS)   // 4096 rows in the projection GEMMs

typedef __bf16 v16bf __attribute__((ext_vector_type(16)));
typedef __bf16 v8bf  __attribute__((ext_vector_type(8)));
typedef float  v8f   __attribute__((ext_vector_type(8)));

__device__ __forceinline__ v8f wmma_bf16(v16bf a, v16bf b, v8f c) {
  // D = A(16x32 bf16) * B(32x16 bf16) + C(16x16 f32)
  return __builtin_amdgcn_wmma_f32_16x16x32_bf16(
      /*neg_a=*/false, a, /*neg_b=*/false, b,
      /*c_mod=*/(short)0, c, /*reuse_a=*/false, /*reuse_b=*/false);
}

// ---- CDNA5 async global->LDS copy (ASYNCcnt-tracked, no VGPR bounce) ----
// LDS offset = low 32 bits of the generic pointer (shared aperture: low
// 32 bits are the LDS byte offset).
__device__ __forceinline__ void async_ld_b128(void* lds_ptr, const void* gptr) {
  unsigned lds_off = (unsigned)(size_t)lds_ptr;
  asm volatile("global_load_async_to_lds_b128 %0, %1, off"
               :: "v"(lds_off), "v"(gptr) : "memory");
}
__device__ __forceinline__ void wait_async0() {
  asm volatile("s_wait_asynccnt 0" ::: "memory");
}
__device__ __forceinline__ void lds_fence() {
  asm volatile("s_wait_dscnt 0" ::: "memory");
}

// A fragment 16(M)x32(K): lane m (0-15) holds K 0-7 & 16-23, lane m+16 holds
// K 8-15 & 24-31 (ISA 7.12.2, 16-bit A layout). Tile is [m][k], stride ldk.
__device__ __forceinline__ v16bf load_a_frag(const __bf16* t, int ldk) {
  int lane = threadIdx.x & 31;
  int m = lane & 15, hi = lane >> 4;
  union { v16bf v; v8bf h[2]; } u;
  const __bf16* p = t + m * ldk + hi * 8;
  u.h[0] = *(const v8bf*)(p);        // K = hi*8 .. hi*8+7
  u.h[1] = *(const v8bf*)(p + 16);   // K = 16+hi*8 ..
  return u.v;
}

// B fragment 32(K)x16(N): lanes 0-15 hold K 0-7 & 16-23 of column N=lane,
// lanes 16-31 hold K 8-15 & 24-31 of column N=lane-16. Tile stored column-major
// [n][k] (column n contiguous along k), stride ldk.
__device__ __forceinline__ v16bf load_b_frag(const __bf16* t, int ldk) {
  int lane = threadIdx.x & 31;
  int n = lane & 15, hi = lane >> 4;
  union { v16bf v; v8bf h[2]; } u;
  const __bf16* p = t + n * ldk + hi * 8;
  u.h[0] = *(const v8bf*)(p);
  u.h[1] = *(const v8bf*)(p + 16);
  return u.v;
}

// reductions across the 16-lane halves of a wave32 (rows of the C fragment)
__device__ __forceinline__ float redmax16(float v) {
  v = fmaxf(v, __shfl_xor(v, 1, 32));
  v = fmaxf(v, __shfl_xor(v, 2, 32));
  v = fmaxf(v, __shfl_xor(v, 4, 32));
  v = fmaxf(v, __shfl_xor(v, 8, 32));
  return v;
}
__device__ __forceinline__ float redsum16(float v) {
  v += __shfl_xor(v, 1, 32);
  v += __shfl_xor(v, 2, 32);
  v += __shfl_xor(v, 4, 32);
  v += __shfl_xor(v, 8, 32);
  return v;
}

// ---------------- fp32 -> bf16 convert (vectorized) ----------------
__global__ void cvt_bf16_v4(const float4* __restrict__ in,
                            __bf16* __restrict__ out, int n4) {
  int i = blockIdx.x * 256 + threadIdx.x;
  if (i < n4) {
    float4 f = in[i];
    union { unsigned long long u; __bf16 e[4]; } o;
    o.e[0] = (__bf16)f.x; o.e[1] = (__bf16)f.y;
    o.e[2] = (__bf16)f.z; o.e[3] = (__bf16)f.w;
    *(unsigned long long*)&out[(size_t)i * 4] = o.u;
  }
}

// ---------------- projection GEMM: out = A @ W^T + bias ----------------
// Workgroup tile 64M x 64N (4 waves; wave w owns rows w*16..w*16+15 and all
// 64 N columns -> one A-frag reused across 4 B-frags / 4 WMMAs per k-step).
// A: [MM, EE] bf16 row-major. W: [EE, EE] bf16 row-major (row n = weights of
// output feature n, i.e. B column n = W row n -> contiguous).
// SPLIT: write bf16 to head-split layout [B,H,S,D]; else write [M,N]
// (fp32 if F32OUT, used for the final projection straight into d_out).
template <bool SPLIT, bool F32OUT>
__global__ __launch_bounds__(128) void proj_gemm(
    const __bf16* __restrict__ A, const __bf16* __restrict__ W,
    const float* __restrict__ bias, void* __restrict__ outp) {
  __shared__ __bf16 At[64 * 40];   // [m][k] padded: 80B rows -> conflict-free
  __shared__ __bf16 Wt[64 * 40];   // [n][k] padded
  const int tid = threadIdx.x;           // 0..127
  const int lane = tid & 31, wv = tid >> 5;
  const int mbase = blockIdx.x * 64;
  const int nbase = blockIdx.y * 64;

  v8f c0 = {}, c1 = {}, c2 = {}, c3 = {};
  const int row = tid >> 1, col = (tid & 1) * 16;   // tile-copy assignment
  for (int k0 = 0; k0 < EE; k0 += 32) {
    __syncthreads();                 // prev-iter frag reads done in all waves
    // A tile 64x32 and W tile 64x32 via async global->LDS (2x16B each)
    {
      const __bf16* ga = &A[(size_t)(mbase + row) * EE + k0 + col];
      async_ld_b128(&At[row * 40 + col],     ga);
      async_ld_b128(&At[row * 40 + col + 8], ga + 8);
      const __bf16* gw = &W[(size_t)(nbase + row) * EE + k0 + col];
      async_ld_b128(&Wt[row * 40 + col],     gw);
      async_ld_b128(&Wt[row * 40 + col + 8], gw + 8);
    }
    wait_async0();
    __syncthreads();
    v16bf af = load_a_frag(At + wv * 16 * 40, 40);
    c0 = wmma_bf16(af, load_b_frag(Wt,           40), c0);
    c1 = wmma_bf16(af, load_b_frag(Wt + 16 * 40, 40), c1);
    c2 = wmma_bf16(af, load_b_frag(Wt + 32 * 40, 40), c2);
    c3 = wmma_bf16(af, load_b_frag(Wt + 48 * 40, 40), c3);
  }

  // epilogue: C layout -> VGPR r: lanes 0-15 row r, lanes 16-31 row r+8
  const int hi = lane >> 4, nc = lane & 15;
  v8f* cs[4] = {&c0, &c1, &c2, &c3};
  for (int n = 0; n < 4; ++n) {
    for (int r = 0; r < 8; ++r) {
      int mrow = mbase + wv * 16 + r + hi * 8;
      int ncol = nbase + n * 16 + nc;
      float v = (*cs[n])[r] + bias[ncol];
      if (SPLIT) {
        int b = mrow >> 11, s = mrow & (SS - 1);
        int h = ncol >> 6, d = ncol & 63;
        ((__bf16*)outp)[(((size_t)(b * HH + h)) * SS + s) * DD + d] = (__bf16)v;
      } else if (F32OUT) {
        ((float*)outp)[(size_t)mrow * EE + ncol] = v;
      } else {
        ((__bf16*)outp)[(size_t)mrow * EE + ncol] = (__bf16)v;
      }
    }
  }
}

// ---------------- flash attention (causal + key padding mask) ------------
// Q/K/V: [B,H,S,D] bf16.  attnout: [B,S,E] bf16 (heads re-concatenated).
// One WG = 4 waves = 64 query rows of one (b,h); wave w owns 16 rows.
__global__ __launch_bounds__(128) void flash_attn(
    const __bf16* __restrict__ Qh, const __bf16* __restrict__ Kh,
    const __bf16* __restrict__ Vh, const unsigned char* __restrict__ maskp,
    __bf16* __restrict__ attnout) {
  __shared__ __bf16 Kt[32 * 72];      // [key][d], stride 72 -> 144B rows
  __shared__ __bf16 Vt[64 * 40];      // transposed [d][key], stride 40
  __shared__ __bf16 Qs[4][16 * 72];   // per-wave Q staging
  __shared__ __bf16 Ps[4][16 * 40];   // per-wave P (C-layout -> A-layout)

  const int tid = threadIdx.x, lane = tid & 31, wv = tid >> 5;
  const int tilesPerBH = SS / 64;                 // 32
  const int bh = blockIdx.x / tilesPerBH;
  const int qt = blockIdx.x % tilesPerBH;
  const int b = bh / HH, h = bh % HH;
  const int qwg = qt * 64;
  const int qwave = qwg + wv * 16;
  const __bf16* Qb = Qh + (size_t)bh * SS * DD;
  const __bf16* Kb = Kh + (size_t)bh * SS * DD;
  const __bf16* Vb = Vh + (size_t)bh * SS * DD;
  const unsigned char* mrow = maskp + (size_t)b * SS;

  // stage this wave's 16x64 Q tile (async), then pull the two A fragments
  for (int p = 0; p < 4; ++p) {
    int cidx = p * 32 + lane;
    int row = cidx >> 3, col = (cidx & 7) * 8;
    async_ld_b128(&Qs[wv][row * 72 + col],
                  &Qb[(size_t)(qwave + row) * DD + col]);
  }
  wait_async0();
  v16bf aq0 = load_a_frag(&Qs[wv][0], 72);    // d 0..31
  v16bf aq1 = load_a_frag(&Qs[wv][32], 72);   // d 32..63

  float m_[8], l_[8];
  v8f o0 = {}, o1 = {}, o2 = {}, o3 = {};
  for (int r = 0; r < 8; ++r) { m_[r] = -3.0e38f; l_[r] = 0.0f; }

  const int hi = lane >> 4, nc = lane & 15;
  const int nblocks = qwg / 32 + 2;            // uniform per WG (causal bound)
  const int qmaxw = qwave + 15;

  for (int j = 0; j < nblocks; ++j) {
    const int kbase = j * 32;
    __syncthreads();
    // K tile [32][64] -> Kt via async LDS DMA (B columns for Q.K^T)
    for (int p = 0; p < 2; ++p) {
      int cidx = p * 128 + tid;
      int row = cidx >> 3, col = (cidx & 7) * 8;
      async_ld_b128(&Kt[row * 72 + col],
                    &Kb[(size_t)(kbase + row) * DD + col]);
    }
    // V tile [32][64] -> transposed Vt[d][key] (scatter; async can't transpose)
    for (int p = 0; p < 2; ++p) {
      int cidx = p * 128 + tid;
      int row = cidx >> 3, col = (cidx & 7) * 8;
      union { uint4 u; __bf16 e[8]; } vvu;
      vvu.u = *(const uint4*)&Vb[(size_t)(kbase + row) * DD + col];
      for (int jj = 0; jj < 8; ++jj) Vt[(col + jj) * 40 + row] = vvu.e[jj];
    }
    wait_async0();
    __syncthreads();

    if (kbase <= qmaxw) {                      // wave-uniform causal skip
      v16bf bk0a = load_b_frag(Kt, 72);              // keys 0-15, d 0-31
      v16bf bk0b = load_b_frag(Kt + 32, 72);         // keys 0-15, d 32-63
      v16bf bk1a = load_b_frag(Kt + 16 * 72, 72);    // keys 16-31
      v16bf bk1b = load_b_frag(Kt + 16 * 72 + 32, 72);
      v8f s0 = {}, s1 = {};
      s0 = wmma_bf16(aq0, bk0a, s0);
      s0 = wmma_bf16(aq1, bk0b, s0);
      s1 = wmma_bf16(aq0, bk1a, s1);
      s1 = wmma_bf16(aq1, bk1b, s1);

      const int k0i = kbase + nc, k1i = k0i + 16;
      const bool pad0 = mrow[k0i] != 0, pad1 = mrow[k1i] != 0;
      const float NEGINF = -__builtin_inff();

      float mnew[8], alpha[8];
      for (int r = 0; r < 8; ++r) {
        int q = qwave + r + hi * 8;
        float a  = (pad0 || k0i > q) ? NEGINF : s0[r] * 0.125f;
        float bb = (pad1 || k1i > q) ? NEGINF : s1[r] * 0.125f;
        s0[r] = a; s1[r] = bb;
        float mx = redmax16(fmaxf(a, bb));
        mnew[r] = fmaxf(m_[r], mx);
        alpha[r] = __expf(m_[r] - mnew[r]);    // block 0 always has k=0 valid
      }
      for (int r = 0; r < 8; ++r) {
        float p0 = __expf(s0[r] - mnew[r]);
        float p1 = __expf(s1[r] - mnew[r]);
        l_[r] = l_[r] * alpha[r] + redsum16(p0 + p1);
        m_[r] = mnew[r];
        Ps[wv][(r + hi * 8) * 40 + nc]      = (__bf16)p0;
        Ps[wv][(r + hi * 8) * 40 + nc + 16] = (__bf16)p1;
        o0[r] *= alpha[r]; o1[r] *= alpha[r];
        o2[r] *= alpha[r]; o3[r] *= alpha[r];
      }
      v16bf bv0 = load_b_frag(Vt, 40);
      v16bf bv1 = load_b_frag(Vt + 16 * 40, 40);
      v16bf bv2 = load_b_frag(Vt + 32 * 40, 40);
      v16bf bv3 = load_b_frag(Vt + 48 * 40, 40);
      lds_fence();                             // Ps write -> A-frag read
      v16bf ap = load_a_frag(&Ps[wv][0], 40);
      o0 = wmma_bf16(ap, bv0, o0);
      o1 = wmma_bf16(ap, bv1, o1);
      o2 = wmma_bf16(ap, bv2, o2);
      o3 = wmma_bf16(ap, bv3, o3);
    }
  }

  // epilogue: heads concatenated -> attnout[b, s, h*64 + d] (bf16)
  for (int r = 0; r < 8; ++r) {
    int srow = qwave + r + hi * 8;
    float inv = 1.0f / l_[r];
    size_t base = ((size_t)(b * SS + srow)) * EE + (size_t)h * DD;
    attnout[base + nc]      = (__bf16)(o0[r] * inv);
    attnout[base + 16 + nc] = (__bf16)(o1[r] * inv);
    attnout[base + 32 + nc] = (__bf16)(o2[r] * inv);
    attnout[base + 48 + nc] = (__bf16)(o3[r] * inv);
  }
}

// ---------------- launch ----------------
extern "C" void kernel_launch(void* const* d_in, const int* in_sizes, int n_in,
                              void* d_out, int out_size, void* d_ws,
                              size_t ws_size, hipStream_t stream) {
  const float* x  = (const float*)d_in[0];
  const unsigned char* mask = (const unsigned char*)d_in[1];
  const float* Wq = (const float*)d_in[2];
  const float* bq = (const float*)d_in[3];
  const float* Wk = (const float*)d_in[4];
  const float* bk = (const float*)d_in[5];
  const float* Wv = (const float*)d_in[6];
  const float* bv = (const float*)d_in[7];
  const float* Wo = (const float*)d_in[8];
  const float* bo = (const float*)d_in[9];

  // workspace layout (bf16 elements): 48 MB total
  __bf16* ws  = (__bf16*)d_ws;
  __bf16* xb  = ws;                                   // [MM, EE]
  __bf16* wqb = xb  + (size_t)MM * EE;
  __bf16* wkb = wqb + (size_t)EE * EE;
  __bf16* wvb = wkb + (size_t)EE * EE;
  __bf16* wob = wvb + (size_t)EE * EE;
  __bf16* Qh  = wob + (size_t)EE * EE;                // [B,H,S,D]
  __bf16* Kh  = Qh  + (size_t)BB * HH * SS * DD;
  __bf16* Vh  = Kh  + (size_t)BB * HH * SS * DD;
  __bf16* ao  = Vh  + (size_t)BB * HH * SS * DD;      // [B,S,E]

  // fp32 -> bf16 converts
  cvt_bf16_v4<<<(MM * EE / 4) / 256, 256, 0, stream>>>((const float4*)x,  xb,  MM * EE / 4);
  cvt_bf16_v4<<<(EE * EE / 4) / 256, 256, 0, stream>>>((const float4*)Wq, wqb, EE * EE / 4);
  cvt_bf16_v4<<<(EE * EE / 4) / 256, 256, 0, stream>>>((const float4*)Wk, wkb, EE * EE / 4);
  cvt_bf16_v4<<<(EE * EE / 4) / 256, 256, 0, stream>>>((const float4*)Wv, wvb, EE * EE / 4);
  cvt_bf16_v4<<<(EE * EE / 4) / 256, 256, 0, stream>>>((const float4*)Wo, wob, EE * EE / 4);

  // Q/K/V projections into head-split bf16 layout (WG tile 64x64)
  dim3 pg(MM / 64, EE / 64);
  proj_gemm<true, false><<<pg, 128, 0, stream>>>(xb, wqb, bq, Qh);
  proj_gemm<true, false><<<pg, 128, 0, stream>>>(xb, wkb, bk, Kh);
  proj_gemm<true, false><<<pg, 128, 0, stream>>>(xb, wvb, bv, Vh);

  // flash attention: one WG per (b, h, 64 query rows)
  flash_attn<<<BB * HH * (SS / 64), 128, 0, stream>>>(Qh, Kh, Vh, mask, ao);

  // output projection, fp32 result straight into d_out
  proj_gemm<false, true><<<pg, 128, 0, stream>>>(ao, wob, bo, d_out);
}
